// SentenceEncoder_3470333575170
// MI455X (gfx1250) — matrix operand order repaired and checked by reading, not compiled, not run on hardware
//
#include <hip/hip_runtime.h>

// ---------------- problem constants ----------------
constexpr int S  = 120;
constexpr int T  = 10;
constexpr int B  = 128;
constexpr int E  = 80;
constexpr int HW = 75;
constexpr int HS = 700;
constexpr int NW = S * B;            // 15360 word-level batch rows

constexpr int pad32i(int x) { return (x + 31) & ~31; }
constexpr int pad64i(int x) { return (x + 63) & ~63; }

// ---------------- WMMA types ----------------
typedef __attribute__((ext_vector_type(16))) __bf16        v16bf;
typedef __attribute__((ext_vector_type(8)))  float         v8f;
typedef __attribute__((ext_vector_type(4)))  unsigned int  u32x4;

union Frag { v16bf v; u32x4 q[2]; };

__device__ __forceinline__ unsigned short f2bf(float f) {
  union { float f; unsigned int u; } c;
  c.f = f;
  unsigned int u = c.u;
  unsigned int rnd = ((u >> 16) & 1u) + 0x7FFFu;   // round-to-nearest-even
  return (unsigned short)((u + rnd) >> 16);
}

// ============================================================
// C[M,N] = A[M,Kp] x W[Np,Kp]^T + bias[N]
// A, W are bf16 with K padded to 32 (zeros) and W rows padded to 64 (zeros),
// so the inner loop is completely branch-free; all M here are multiples of 64.
// One wave -> 16(M) x 64(N); 4 waves/block -> 64x64 macro tile.
// Steady state: clauses of global_load_b128 + 4x v_wmma_f32_16x16x32_bf16.
// ============================================================
__global__ void __launch_bounds__(128)
wmma_gemm_bf16(const unsigned short* __restrict__ A, int lda,
               const unsigned short* __restrict__ W, int ldw,
               const float* __restrict__ bias,
               float* __restrict__ C, int ldc,
               int N, int K32) {
  const int lane  = threadIdx.x & 31;
  const int wave  = threadIdx.x >> 5;
  const int mBase = (blockIdx.x * 4 + wave) * 16;
  const int nBase = blockIdx.y * 64;
  const int half  = lane >> 4;          // 0: lanes 0-15, 1: lanes 16-31
  const int nCol  = lane & 15;

  // A fragment (16x32 bf16): lane covers 16 consecutive k at k = half*16
  const unsigned short* Ap = A + (size_t)(mBase + nCol) * lda + half * 16;
  // B fragment: lane covers 16 consecutive k at k = half*16; n-tiles 16 rows apart
  const unsigned short* Wp = W + (size_t)(nBase + nCol) * ldw + half * 16;
  const size_t wtile = (size_t)16 * ldw;

  v8f acc0 = {}; v8f acc1 = {}; v8f acc2 = {}; v8f acc3 = {};

  for (int c = 0; c < K32; ++c) {
    Frag a, b0, b1, b2, b3;
    a.q[0]  = *(const u32x4*)(Ap);
    a.q[1]  = *(const u32x4*)(Ap + 8);
    b0.q[0] = *(const u32x4*)(Wp);
    b0.q[1] = *(const u32x4*)(Wp + 8);
    b1.q[0] = *(const u32x4*)(Wp + wtile);
    b1.q[1] = *(const u32x4*)(Wp + wtile + 8);
    b2.q[0] = *(const u32x4*)(Wp + 2 * wtile);
    b2.q[1] = *(const u32x4*)(Wp + 2 * wtile + 8);
    b3.q[0] = *(const u32x4*)(Wp + 3 * wtile);
    b3.q[1] = *(const u32x4*)(Wp + 3 * wtile + 8);
    __builtin_prefetch(Ap + 32, 0, 3);          // WGP-scope speculative prefetch
    __builtin_prefetch(Wp + 32, 0, 3);
    acc0 = __builtin_amdgcn_wmma_f32_16x16x32_bf16(false, a.v, false, b0.v, (short)0, acc0, false, false);
    acc1 = __builtin_amdgcn_wmma_f32_16x16x32_bf16(false, a.v, false, b1.v, (short)0, acc1, false, false);
    acc2 = __builtin_amdgcn_wmma_f32_16x16x32_bf16(false, a.v, false, b2.v, (short)0, acc2, false, false);
    acc3 = __builtin_amdgcn_wmma_f32_16x16x32_bf16(false, a.v, false, b3.v, (short)0, acc3, false, false);
    Ap += 32;
    Wp += 32;
  }

#define STORE_TILE(ACC, NT)                                          \
  do {                                                               \
    int n = nBase + (NT) * 16 + nCol;                                \
    if (n < N) {                                                     \
      float bv = bias[n];                                            \
      _Pragma("unroll")                                              \
      for (int r = 0; r < 8; ++r) {                                  \
        int m = mBase + (half << 3) + r;                             \
        C[(size_t)m * ldc + n] = ACC[r] + bv;                        \
      }                                                              \
    }                                                                \
  } while (0)
  STORE_TILE(acc0, 0); STORE_TILE(acc1, 1);
  STORE_TILE(acc2, 2); STORE_TILE(acc3, 3);
#undef STORE_TILE
}

// ============================================================
// Elementwise GRU gate math (PyTorch gate order r,z,n).
// Writes fp32 h (for next step's z*h) plus bf16 h (padded stride, next GEMM),
// and optionally the bf16 layer output y (pre-offset pointer).
// ============================================================
__global__ void gru_step_kernel(const float* __restrict__ xw_t,   // [N,3H]
                                const float* __restrict__ gh,     // [N,3H]
                                float* __restrict__ h,            // [N,H]
                                unsigned short* __restrict__ hb,  // [N,Hpad]
                                int hbStride,
                                unsigned short* __restrict__ y,   // optional
                                int ystride, int Nrows, int H) {
  int idx = blockIdx.x * blockDim.x + threadIdx.x;
  if (idx >= Nrows * H) return;
  int n = idx / H;
  int j = idx - n * H;
  const float* xr = xw_t + (size_t)n * 3 * H;
  const float* gr = gh   + (size_t)n * 3 * H;
  float r  = 1.0f / (1.0f + __expf(-(xr[j] + gr[j])));
  float z  = 1.0f / (1.0f + __expf(-(xr[H + j] + gr[H + j])));
  float nn = tanhf(xr[2 * H + j] + r * gr[2 * H + j]);
  float hn = (1.0f - z) * nn + z * h[idx];
  h[idx] = hn;
  unsigned short hv = f2bf(hn);
  hb[(size_t)n * hbStride + j] = hv;
  if (y) y[(size_t)n * ystride + j] = hv;
}

// xe_bf[t, s*B+b, e] (bf16, padded cols) = emb[x[s,t,b], e]
__global__ void embed_bf16_kernel(const int* __restrict__ x,
                                  const float* __restrict__ emb,
                                  unsigned short* __restrict__ xe,
                                  int colsPad) {
  size_t idx = (size_t)blockIdx.x * blockDim.x + threadIdx.x;
  size_t total = (size_t)T * NW * colsPad;
  if (idx >= total) return;
  int e = (int)(idx % colsPad);
  size_t r = idx / colsPad;           // t*S*B + s*B + b
  int b = (int)(r % B);
  size_t r2 = r / B;
  int s = (int)(r2 % S);
  int t = (int)(r2 / S);
  unsigned short v = 0;
  if (e < E) {
    int tok = x[((size_t)s * T + t) * B + b];
    v = f2bf(emb[(size_t)tok * E + e]);
  }
  xe[idx] = v;
}

// fp32 [rows,cols] -> bf16 [rowsPad, colsPad], zero padded
__global__ void cvt_pad_kernel(const float* __restrict__ src, int rows, int cols,
                               unsigned short* __restrict__ dst,
                               int rowsPad, int colsPad) {
  size_t idx = (size_t)blockIdx.x * blockDim.x + threadIdx.x;
  size_t total = (size_t)rowsPad * colsPad;
  if (idx >= total) return;
  int c = (int)(idx % colsPad);
  int r = (int)(idx / colsPad);
  dst[idx] = (r < rows && c < cols) ? f2bf(src[(size_t)r * cols + c]) : (unsigned short)0;
}

__global__ void fill_zero_f32(float* __restrict__ p, size_t n) {
  size_t idx = (size_t)blockIdx.x * blockDim.x + threadIdx.x;
  if (idx < n) p[idx] = 0.0f;
}
__global__ void fill_zero_u16(unsigned short* __restrict__ p, size_t n) {
  size_t idx = (size_t)blockIdx.x * blockDim.x + threadIdx.x;
  if (idx < n) p[idx] = 0;
}

// h fp32 [N,H] -> bf16 column slice of dst
__global__ void copy_cols_bf16_kernel(const float* __restrict__ src, int H,
                                      unsigned short* __restrict__ dst,
                                      int dstride, int doff, int Nrows) {
  int idx = blockIdx.x * blockDim.x + threadIdx.x;
  if (idx >= Nrows * H) return;
  int n = idx / H;
  int j = idx - n * H;
  dst[(size_t)n * dstride + doff + j] = f2bf(src[idx]);
}
// h fp32 [N,H] -> fp32 column slice of dst (final output)
__global__ void copy_cols_f32_kernel(const float* __restrict__ src, int H,
                                     float* __restrict__ dst,
                                     int dstride, int doff, int Nrows) {
  int idx = blockIdx.x * blockDim.x + threadIdx.x;
  if (idx >= Nrows * H) return;
  int n = idx / H;
  int j = idx - n * H;
  dst[(size_t)n * dstride + doff + j] = src[idx];
}

// ---------------- host-side helpers ----------------
static inline void launch_gemm(const unsigned short* A, int lda,
                               const unsigned short* W, int ldw,
                               const float* bias, float* C, int ldc,
                               int M, int N, int K32, hipStream_t stream) {
  dim3 grid((unsigned)(M / 64), (unsigned)(pad64i(N) / 64));
  wmma_gemm_bf16<<<grid, dim3(128), 0, stream>>>(A, lda, W, ldw, bias, C, ldc, N, K32);
}

static inline unsigned short* cvt_weight(const float* src, int rows, int cols,
                                         unsigned short* dst, hipStream_t stream) {
  int rp = pad64i(rows), cp = pad32i(cols);
  size_t total = (size_t)rp * cp;
  cvt_pad_kernel<<<(unsigned)((total + 255) / 256), 256, 0, stream>>>(
      src, rows, cols, dst, rp, cp);
  return dst + total;
}

static inline void zero_f32(float* p, size_t n, hipStream_t s) {
  if (n) fill_zero_f32<<<(unsigned)((n + 255) / 256), 256, 0, s>>>(p, n);
}
static inline void zero_u16(unsigned short* p, size_t n, hipStream_t s) {
  if (n) fill_zero_u16<<<(unsigned)((n + 255) / 256), 256, 0, s>>>(p, n);
}

// One directional GRU scan. xw = precomputed input gates [Tlen, Nrows, 3H] (fp32).
static void run_scan(const float* xw, int Tlen, int Nrows, int H,
                     const unsigned short* WhhB, const float* bhh,
                     float* h, unsigned short* hb, float* gh,
                     unsigned short* y, int ystride, int yoff,
                     bool backward, hipStream_t stream) {
  int Hp = pad32i(H);
  int total = Nrows * H;
  zero_f32(h, (size_t)total, stream);
  zero_u16(hb, (size_t)Nrows * Hp, stream);
  for (int i = 0; i < Tlen; ++i) {
    int t = backward ? (Tlen - 1 - i) : i;
    // gh = h @ Whh^T + bhh   (recurrent WMMA GEMM)
    launch_gemm(hb, Hp, WhhB, Hp, bhh, gh, 3 * H, Nrows, 3 * H, Hp / 32, stream);
    const float* xw_t = xw + (size_t)t * Nrows * 3 * H;
    unsigned short* y_t = y ? (y + (size_t)t * Nrows * ystride + yoff) : nullptr;
    gru_step_kernel<<<(total + 255) / 256, 256, 0, stream>>>(
        xw_t, gh, h, hb, Hp, y_t, ystride, Nrows, H);
  }
}

// ---------------- workspace layout ----------------
// fp32 region (elements)
constexpr size_t SZ_XW = (size_t)T * NW * 3 * HW;        // 34,560,000
constexpr size_t SZ_GH = (size_t)NW * 3 * HW;            //  3,456,000
constexpr size_t SZ_H  = (size_t)NW * HW;                //  1,152,000
// bf16 region (elements)
constexpr int EP   = pad32i(E);        // 96
constexpr int Y0P  = pad32i(2 * HW);   // 160
constexpr int WDP  = pad32i(4 * HW);   // 320
constexpr int Y1P  = pad32i(2 * HS);   // 1408
constexpr int HWP  = pad32i(HW);       // 96
constexpr int HSP  = pad32i(HS);       // 704
constexpr size_t US_XE    = (size_t)T * NW * EP;                          // 14,745,600
constexpr size_t US_Y     = (size_t)T * NW * Y0P;                         // 24,576,000
constexpr size_t US_WORDS = (size_t)NW * WDP;                             //  4,915,200
constexpr size_t US_HB    = (size_t)NW * HWP;                             //  1,474,560
constexpr size_t WPD(int r, int c) { return (size_t)pad64i(r) * pad32i(c); }
constexpr size_t US_WTS =
    2 * (WPD(3 * HW, E) + WPD(3 * HW, HW) + WPD(3 * HW, 2 * HW) + WPD(3 * HW, HW)) +
    2 * (WPD(3 * HS, 4 * HW) + WPD(3 * HS, HS) + WPD(3 * HS, 2 * HS) + WPD(3 * HS, HS));

static_assert((size_t)S * B * 3 * HS <= SZ_XW, "xw reuse");
static_assert((size_t)S * B * Y1P <= US_Y, "y reuse");
static_assert((size_t)B * 3 * HS <= SZ_GH, "gh reuse");
static_assert((size_t)B * HS <= SZ_H, "h reuse");
static_assert((size_t)B * HSP <= US_HB, "hb reuse");

extern "C" void kernel_launch(void* const* d_in, const int* in_sizes, int n_in,
                              void* d_out, int out_size, void* d_ws, size_t ws_size,
                              hipStream_t stream) {
  (void)in_sizes; (void)n_in; (void)out_size; (void)ws_size;

  const int*   x       = (const int*)d_in[0];
  const float* emb     = (const float*)d_in[1];
  const float* we_Wih0 = (const float*)d_in[2];
  const float* we_Whh0 = (const float*)d_in[3];
  const float* we_bih0 = (const float*)d_in[4];
  const float* we_bhh0 = (const float*)d_in[5];
  const float* we_Wih1 = (const float*)d_in[6];
  const float* we_Whh1 = (const float*)d_in[7];
  const float* we_bih1 = (const float*)d_in[8];
  const float* we_bhh1 = (const float*)d_in[9];
  const float* se_Wih0 = (const float*)d_in[10];
  const float* se_Whh0 = (const float*)d_in[11];
  const float* se_bih0 = (const float*)d_in[12];
  const float* se_bhh0 = (const float*)d_in[13];
  const float* se_Wih1 = (const float*)d_in[14];
  const float* se_Whh1 = (const float*)d_in[15];
  const float* se_bih1 = (const float*)d_in[16];
  const float* se_bhh1 = (const float*)d_in[17];
  float* out = (float*)d_out;               // [1, B, 4*HS]

  // ---- carve workspace ----
  float* fp = (float*)d_ws;
  float* xwA   = fp;            fp += SZ_XW;
  float* xwB   = fp;            fp += SZ_XW;
  float* ghbuf = fp;            fp += SZ_GH;
  float* hbuf  = fp;            fp += SZ_H;
  unsigned short* us = (unsigned short*)fp;
  unsigned short* xe_bf    = us;  us += US_XE;
  unsigned short* y_bf     = us;  us += US_Y;
  unsigned short* words_bf = us;  us += US_WORDS;
  unsigned short* h_bf     = us;  us += US_HB;
  unsigned short* wts      = us;  // US_WTS elements

  // ---- one-time bf16 weight conversion (padded, zero-filled) ----
  unsigned short* p = wts;
  unsigned short* wWih0f = p; p = cvt_weight(we_Wih0,                    3 * HW, E,      p, stream);
  unsigned short* wWih0b = p; p = cvt_weight(we_Wih0 + 3 * HW * E,       3 * HW, E,      p, stream);
  unsigned short* wWhh0f = p; p = cvt_weight(we_Whh0,                    3 * HW, HW,     p, stream);
  unsigned short* wWhh0b = p; p = cvt_weight(we_Whh0 + 3 * HW * HW,      3 * HW, HW,     p, stream);
  unsigned short* wWih1f = p; p = cvt_weight(we_Wih1,                    3 * HW, 2 * HW, p, stream);
  unsigned short* wWih1b = p; p = cvt_weight(we_Wih1 + 3 * HW * 2 * HW,  3 * HW, 2 * HW, p, stream);
  unsigned short* wWhh1f = p; p = cvt_weight(we_Whh1,                    3 * HW, HW,     p, stream);
  unsigned short* wWhh1b = p; p = cvt_weight(we_Whh1 + 3 * HW * HW,      3 * HW, HW,     p, stream);
  unsigned short* sWih0f = p; p = cvt_weight(se_Wih0,                    3 * HS, 4 * HW, p, stream);
  unsigned short* sWih0b = p; p = cvt_weight(se_Wih0 + 3 * HS * 4 * HW,  3 * HS, 4 * HW, p, stream);
  unsigned short* sWhh0f = p; p = cvt_weight(se_Whh0,                    3 * HS, HS,     p, stream);
  unsigned short* sWhh0b = p; p = cvt_weight(se_Whh0 + 3 * HS * HS,      3 * HS, HS,     p, stream);
  unsigned short* sWih1f = p; p = cvt_weight(se_Wih1,                    3 * HS, 2 * HS, p, stream);
  unsigned short* sWih1b = p; p = cvt_weight(se_Wih1 + 3 * HS * 2 * HS,  3 * HS, 2 * HS, p, stream);
  unsigned short* sWhh1f = p; p = cvt_weight(se_Whh1,                    3 * HS, HS,     p, stream);
  unsigned short* sWhh1b = p; p = cvt_weight(se_Whh1 + 3 * HS * HS,      3 * HS, HS,     p, stream);

  // ---- embedding gather (bf16, padded) ----
  embed_bf16_kernel<<<(unsigned)(((size_t)T * NW * EP + 255) / 256), 256, 0, stream>>>(
      x, emb, xe_bf, EP);

  // ================= word encoder, layer 0 (in=E) =================
  launch_gemm(xe_bf, EP, wWih0f, EP, we_bih0,          xwA, 3 * HW, T * NW, 3 * HW, EP / 32, stream);
  launch_gemm(xe_bf, EP, wWih0b, EP, we_bih0 + 3 * HW, xwB, 3 * HW, T * NW, 3 * HW, EP / 32, stream);
  zero_u16(y_bf, (size_t)T * NW * Y0P, stream);     // pad cols of y0 must be 0
  zero_u16(words_bf, US_WORDS, stream);             // pad cols (300..319) of words must be 0
  run_scan(xwA, T, NW, HW, wWhh0f, we_bhh0,          hbuf, h_bf, ghbuf, y_bf, Y0P, 0,  false, stream);
  copy_cols_bf16_kernel<<<(NW * HW + 255) / 256, 256, 0, stream>>>(hbuf, HW, words_bf, WDP, 0, NW);
  run_scan(xwB, T, NW, HW, wWhh0b, we_bhh0 + 3 * HW, hbuf, h_bf, ghbuf, y_bf, Y0P, HW, true,  stream);
  copy_cols_bf16_kernel<<<(NW * HW + 255) / 256, 256, 0, stream>>>(hbuf, HW, words_bf, WDP, HW, NW);

  // ================= word encoder, layer 1 (in=2*HW) =================
  launch_gemm(y_bf, Y0P, wWih1f, Y0P, we_bih1,          xwA, 3 * HW, T * NW, 3 * HW, Y0P / 32, stream);
  launch_gemm(y_bf, Y0P, wWih1b, Y0P, we_bih1 + 3 * HW, xwB, 3 * HW, T * NW, 3 * HW, Y0P / 32, stream);
  run_scan(xwA, T, NW, HW, wWhh1f, we_bhh1,          hbuf, h_bf, ghbuf, nullptr, 0, 0, false, stream);
  copy_cols_bf16_kernel<<<(NW * HW + 255) / 256, 256, 0, stream>>>(hbuf, HW, words_bf, WDP, 2 * HW, NW);
  run_scan(xwB, T, NW, HW, wWhh1b, we_bhh1 + 3 * HW, hbuf, h_bf, ghbuf, nullptr, 0, 0, true,  stream);
  copy_cols_bf16_kernel<<<(NW * HW + 255) / 256, 256, 0, stream>>>(hbuf, HW, words_bf, WDP, 3 * HW, NW);

  // ================= sentence encoder, layer 0 (in=4*HW) =================
  launch_gemm(words_bf, WDP, sWih0f, WDP, se_bih0,          xwA, 3 * HS, S * B, 3 * HS, WDP / 32, stream);
  launch_gemm(words_bf, WDP, sWih0b, WDP, se_bih0 + 3 * HS, xwB, 3 * HS, S * B, 3 * HS, WDP / 32, stream);
  zero_u16(y_bf, (size_t)S * B * Y1P, stream);      // re-zero for sentence layout pads
  run_scan(xwA, S, B, HS, sWhh0f, se_bhh0,          hbuf, h_bf, ghbuf, y_bf, Y1P, 0,  false, stream);
  copy_cols_f32_kernel<<<(B * HS + 255) / 256, 256, 0, stream>>>(hbuf, HS, out, 4 * HS, 0, B);
  run_scan(xwB, S, B, HS, sWhh0b, se_bhh0 + 3 * HS, hbuf, h_bf, ghbuf, y_bf, Y1P, HS, true,  stream);
  copy_cols_f32_kernel<<<(B * HS + 255) / 256, 256, 0, stream>>>(hbuf, HS, out, 4 * HS, HS, B);

  // ================= sentence encoder, layer 1 (in=2*HS) =================
  launch_gemm(y_bf, Y1P, sWih1f, Y1P, se_bih1,          xwA, 3 * HS, S * B, 3 * HS, Y1P / 32, stream);
  launch_gemm(y_bf, Y1P, sWih1b, Y1P, se_bih1 + 3 * HS, xwB, 3 * HS, S * B, 3 * HS, Y1P / 32, stream);
  run_scan(xwA, S, B, HS, sWhh1f, se_bhh1,          hbuf, h_bf, ghbuf, nullptr, 0, 0, false, stream);
  copy_cols_f32_kernel<<<(B * HS + 255) / 256, 256, 0, stream>>>(hbuf, HS, out, 4 * HS, 2 * HS, B);
  run_scan(xwB, S, B, HS, sWhh1b, se_bhh1 + 3 * HS, hbuf, h_bf, ghbuf, nullptr, 0, 0, true,  stream);
  copy_cols_f32_kernel<<<(B * HS + 255) / 256, 256, 0, stream>>>(hbuf, HS, out, 4 * HS, 3 * HS, B);
}
